// LongcatFlashMoE_68101001445531
// MI455X (gfx1250) — compile-verified, matrix operand order checked by reading, not run on hardware
//
#include <hip/hip_runtime.h>
#include <hip/hip_bf16.h>
#include <math.h>

#define T_TOK 512
#define HID   1024
#define INTER 512
#define NR    32
#define ETOT  64
#define TOPK  8

typedef __attribute__((ext_vector_type(16))) __bf16 v16bf;
typedef __attribute__((ext_vector_type(4)))  __bf16 v4bf;
typedef __attribute__((ext_vector_type(2)))  __bf16 v2bf;
typedef __attribute__((ext_vector_type(8)))  float  v8f;

// packed fp32x2 -> bf16x2 (v_cvt_pk_bf16_f32 when the builtin exists)
#if __has_builtin(__builtin_amdgcn_cvt_pk_bf16_f32)
__device__ __forceinline__ v2bf cvt2(float x, float y) {
  return __builtin_amdgcn_cvt_pk_bf16_f32(x, y);
}
#else
__device__ __forceinline__ v2bf cvt2(float x, float y) {
  v2bf r; r[0] = (__bf16)x; r[1] = (__bf16)y; return r;
}
#endif

__device__ __forceinline__ v4bf cvt4(float4 v) {
  struct P { v2bf a, b; } p;
  p.a = cvt2(v.x, v.y);
  p.b = cvt2(v.z, v.w);
  return __builtin_bit_cast(v4bf, p);
}

// ---------------------------------------------------------------------------
// Kernel 1: router. One block (64 threads) per token.
// The reference's zero-slot "reassignment" gives reassigned slots weight 0,
// so it cannot affect the output; only cw (normal slots) and the total
// zero-expert weight matter.
// ---------------------------------------------------------------------------
__global__ __launch_bounds__(64)
void moe_router_kernel(const float* __restrict__ hidden,
                       const float* __restrict__ rw,
                       const float* __restrict__ rb,
                       float* __restrict__ cw,
                       float* __restrict__ zscale) {
  __shared__ float hs[HID];
  __shared__ float sc[ETOT];
  const int t   = blockIdx.x;
  const int tid = threadIdx.x;

  for (int c = tid; c < HID; c += 64) hs[c] = hidden[(size_t)t * HID + c];
  __syncthreads();

  const float* wrow = rw + (size_t)tid * HID;
  float acc = 0.f;
  for (int h = 0; h < HID; h += 4) {
    acc = fmaf(hs[h + 0], wrow[h + 0], acc);
    acc = fmaf(hs[h + 1], wrow[h + 1], acc);
    acc = fmaf(hs[h + 2], wrow[h + 2], acc);
    acc = fmaf(hs[h + 3], wrow[h + 3], acc);
  }
  sc[tid] = acc;
  __syncthreads();

  if (tid == 0) {  // serial softmax + top-8 over 64 (deterministic tie order)
    float mx = sc[0];
    for (int i = 1; i < ETOT; ++i) mx = fmaxf(mx, sc[i]);
    float p[ETOT];
    float sum = 0.f;
    for (int i = 0; i < ETOT; ++i) { p[i] = __expf(sc[i] - mx); sum += p[i]; }
    const float inv = 1.f / sum;

    unsigned long long taken = 0ull;
    float cwl[NR];
    for (int j = 0; j < NR; ++j) cwl[j] = 0.f;
    float zw = 0.f;
    for (int k = 0; k < TOPK; ++k) {
      int bi = 0; float bv = -INFINITY;
      for (int i = 0; i < ETOT; ++i) {
        if ((taken >> i) & 1ull) continue;
        const float v = p[i] * inv + rb[i];   // bias only for selection
        if (v > bv) { bv = v; bi = i; }
      }
      taken |= 1ull << bi;
      const float w = p[bi] * inv;            // ROUTED_SCALING == 1.0
      if (bi >= NR) zw += w; else cwl[bi] += w;
    }
    for (int j = 0; j < NR; ++j) cw[t * NR + j] = cwl[j];
    zscale[t] = zw;
  }
}

// ---------------------------------------------------------------------------
// Kernel 2: deterministic per-expert token lists. One block, lane = expert.
// ---------------------------------------------------------------------------
__global__ __launch_bounds__(32)
void moe_build_lists_kernel(const float* __restrict__ cw,
                            int* __restrict__ cnt,
                            int* __restrict__ list) {
  const int e = threadIdx.x;
  int c = 0;
  for (int t = 0; t < T_TOK; ++t)
    if (cw[t * NR + e] != 0.0f) list[e * T_TOK + c++] = t;
  cnt[e] = c;
}

// ---------------------------------------------------------------------------
// Kernel 3: out[t,:] = hidden[t,:] * zscale[t]  (also initializes d_out)
// ---------------------------------------------------------------------------
__global__ __launch_bounds__(256)
void moe_zero_out_kernel(const float* __restrict__ hidden,
                         const float* __restrict__ zscale,
                         float* __restrict__ out) {
  const int t = blockIdx.x;
  const float s = zscale[t];
  for (int c = threadIdx.x; c < HID; c += 256)
    out[(size_t)t * HID + c] = hidden[(size_t)t * HID + c] * s;
}

// ---------------------------------------------------------------------------
// WMMA fragment helpers (CDNA5 16-bit A/B layout, wave32):
//   lane<16  -> row = lane,    K offsets {0..7} and {16..23}
//   lane>=16 -> row = lane-16, K offsets {8..15} and {24..31}
// ---------------------------------------------------------------------------
__device__ __forceinline__ v16bf load_a_frag(const __bf16* __restrict__ ap) {
  v16bf a;
#pragma unroll
  for (int j = 0; j < 8; ++j) { a[j] = ap[j]; a[j + 8] = ap[j + 16]; }
  return a;
}

// gather one B fragment from an fp32 weight row, packed-converting to bf16
__device__ __forceinline__ v16bf load_b_frag(const float* __restrict__ row, int kbase) {
  const float4* p = (const float4*)(row + kbase);  // kbase is 8-float aligned
  struct Q { v4bf q0, q1, q2, q3; } q;
  q.q0 = cvt4(p[0]);
  q.q1 = cvt4(p[1]);
  q.q2 = cvt4(p[4]);
  q.q3 = cvt4(p[5]);
  return __builtin_bit_cast(v16bf, q);
}

// ---------------------------------------------------------------------------
// Kernel 4: grouped silu-gated FFN. Block = (expert e, 32-token tile).
// 256 threads = 8 waves, two 16-row M-tiles so each converted B fragment
// feeds two WMMAs. Stage1: wave w owns gate cols [w*64, w*64+64) and the
// MATCHING val cols, so act = silu(gate)*val fuses in registers.
// Stage2: out = act x w2[e]^T, scaled by cw, atomically accumulated.
// ---------------------------------------------------------------------------
__global__ __launch_bounds__(256)
void moe_ffn_kernel(const float* __restrict__ hidden,
                    const float* __restrict__ w13,
                    const float* __restrict__ w2,
                    const float* __restrict__ cw,
                    const int* __restrict__ cnt,
                    const int* __restrict__ list,
                    float* __restrict__ out) {
  const int e  = blockIdx.x >> 4;        // 16 tiles of 32 tokens per expert
  const int tt = blockIdx.x & 15;
  const int n_tok = cnt[e];
  if (tt * 32 >= n_tok) return;

  __shared__ __bf16 A_lds[32][HID + 8];      // 66 KB
  __shared__ __bf16 act_lds[32][INTER + 8];  // 33 KB
  __shared__ int    tok_s[32];
  __shared__ float  cw_s[32];

  const int tid   = threadIdx.x;
  const int wave  = tid >> 5;
  const int lane  = tid & 31;
  const int rowM  = lane & 15;
  const int khalf = (lane >> 4) << 3;  // K-half offset, also C/D row offset

  if (tid < 32) {
    const int m   = tt * 32 + tid;
    const int tok = (m < n_tok) ? list[e * T_TOK + m] : 0;
    tok_s[tid] = tok;
    cw_s[tid]  = (m < n_tok) ? cw[tok * NR + e] : 0.0f;  // padded rows weigh 0
  }
  __syncthreads();

  {  // stage A tile [32 x 1024] into LDS as bf16 (8 threads per row)
    const int r  = tid >> 3;
    const int c0 = (tid & 7) * 128;
    const float* src = hidden + (size_t)tok_s[r] * HID + c0;
#pragma unroll
    for (int j = 0; j < 128; j += 4)
      *(v4bf*)&A_lds[r][c0 + j] = cvt4(*(const float4*)(src + j));
  }
  __syncthreads();

  // ---------------- Stage 1: up = A x w13[e]^T, fused silu-gate ----------
  v8f accg[2][4] = {};
  v8f accv[2][4] = {};
  for (int kc = 0; kc < HID; kc += 32) {
    const v16bf a0 = load_a_frag(&A_lds[rowM][kc + khalf]);
    const v16bf a1 = load_a_frag(&A_lds[rowM + 16][kc + khalf]);
#pragma unroll
    for (int nt = 0; nt < 4; ++nt) {
      const int ng = wave * 64 + nt * 16 + rowM;                 // gate col
      const v16bf bg = load_b_frag(w13 + ((size_t)e * (2 * INTER) + ng) * HID,
                                   kc + khalf);
      accg[0][nt] = __builtin_amdgcn_wmma_f32_16x16x32_bf16(
          false, a0, false, bg, (short)0, accg[0][nt], false, false);
      accg[1][nt] = __builtin_amdgcn_wmma_f32_16x16x32_bf16(
          false, a1, false, bg, (short)0, accg[1][nt], false, false);
      const int nv = INTER + wave * 64 + nt * 16 + rowM;          // val col
      const v16bf bv = load_b_frag(w13 + ((size_t)e * (2 * INTER) + nv) * HID,
                                   kc + khalf);
      accv[0][nt] = __builtin_amdgcn_wmma_f32_16x16x32_bf16(
          false, a0, false, bv, (short)0, accv[0][nt], false, false);
      accv[1][nt] = __builtin_amdgcn_wmma_f32_16x16x32_bf16(
          false, a1, false, bv, (short)0, accv[1][nt], false, false);
    }
  }

  // act = silu(gate) * val, written to LDS as bf16 (stage-2 A operand)
#pragma unroll
  for (int mt = 0; mt < 2; ++mt) {
#pragma unroll
    for (int nt = 0; nt < 4; ++nt) {
      const int ncol = wave * 64 + nt * 16 + (lane & 15);
#pragma unroll
      for (int v = 0; v < 8; ++v) {
        const float g = accg[mt][nt][v];
        const float x = accv[mt][nt][v];
        const float s = g / (1.0f + __expf(-g));
        const v2bf pk = cvt2(s * x, 0.f);
        act_lds[v + khalf + mt * 16][ncol] = pk[0];
      }
    }
  }
  __syncthreads();

  // ---------------- Stage 2: D = act x w2[e]^T ---------------------------
  v8f acco[2][8] = {};
  for (int kc = 0; kc < INTER; kc += 32) {
    const v16bf a0 = load_a_frag(&act_lds[rowM][kc + khalf]);
    const v16bf a1 = load_a_frag(&act_lds[rowM + 16][kc + khalf]);
#pragma unroll
    for (int nt = 0; nt < 8; ++nt) {
      const int nh = wave * 128 + nt * 16 + rowM;
      const v16bf b = load_b_frag(w2 + ((size_t)e * HID + nh) * INTER,
                                  kc + khalf);
      acco[0][nt] = __builtin_amdgcn_wmma_f32_16x16x32_bf16(
          false, a0, false, b, (short)0, acco[0][nt], false, false);
      acco[1][nt] = __builtin_amdgcn_wmma_f32_16x16x32_bf16(
          false, a1, false, b, (short)0, acco[1][nt], false, false);
    }
  }

  // epilogue: scale rows by cw[token,e], accumulate into out
#pragma unroll
  for (int mt = 0; mt < 2; ++mt) {
#pragma unroll
    for (int nt = 0; nt < 8; ++nt) {
      const int nh = wave * 128 + nt * 16 + (lane & 15);
#pragma unroll
      for (int v = 0; v < 8; ++v) {
        const int m = v + khalf + mt * 16;
        const float val = acco[mt][nt][v] * cw_s[m];
        unsafeAtomicAdd(&out[(size_t)tok_s[m] * HID + nh], val);
      }
    }
  }
}

// ---------------------------------------------------------------------------
extern "C" void kernel_launch(void* const* d_in, const int* in_sizes, int n_in,
                              void* d_out, int out_size, void* d_ws, size_t ws_size,
                              hipStream_t stream) {
  const float* hidden = (const float*)d_in[0];  // [512,1024]
  const float* rw     = (const float*)d_in[1];  // [64,1024]
  const float* rb     = (const float*)d_in[2];  // [64]
  const float* w13    = (const float*)d_in[3];  // [32,1024,1024]
  const float* w2     = (const float*)d_in[4];  // [32,1024,512]
  float* out = (float*)d_out;                   // [512,1024] fp32

  char* ws = (char*)d_ws;
  float* cw     = (float*)(ws);                          // 64 KB
  float* zscale = (float*)(ws + 65536);                  // 2 KB
  int*   cnt    = (int*)  (ws + 65536 + 2048);           // 128 B
  int*   list   = (int*)  (ws + 65536 + 2048 + 128);     // 64 KB

  moe_router_kernel<<<T_TOK, 64, 0, stream>>>(hidden, rw, rb, cw, zscale);
  moe_build_lists_kernel<<<1, NR, 0, stream>>>(cw, cnt, list);
  moe_zero_out_kernel<<<T_TOK, 256, 0, stream>>>(hidden, zscale, out);
  moe_ffn_kernel<<<NR * 16, 256, 0, stream>>>(hidden, w13, w2, cw, cnt, list, out);
}
